// DeepONetCfCDecoder_8821862826625
// MI455X (gfx1250) — matrix operand order repaired and checked by reading, not compiled, not run on hardware
//
#include <hip/hip_runtime.h>
#include <stdint.h>

// ---------------------------------------------------------------------------
// DeepONet CfC decoder for MI455X (gfx1250), compile-only optimized build.
//  - All dense layers via v_wmma_f32_16x16x32_bf16 (wave32 WMMA).
//  - A-tiles staged to LDS with the Tensor Data Mover (tensor_load_to_lds,
//    6-arg clang-23 form), synchronized with s_wait_tensorcnt; fallback
//    cooperative copy if the builtin is unavailable.
//  - Token/K/V projections deduplicated per time bin (T*K=9600 rows instead
//    of N*K=196608): ~20x FLOP reduction vs naive reference.
//  - rel-bias MLP eliminated analytically (LN of a scalar is constant ->
//    constant score shift -> cancelled by softmax).
// ---------------------------------------------------------------------------

#define N_Q    4096
#define KSENS  48
#define TBINS  200
#define DMODEL 384
#define HDIM   256
#define RANKD  128
#define QFEAT  128   // 104 real features padded to 128 for K%32==0

typedef __bf16 bf16t;
typedef bf16t v16bf __attribute__((ext_vector_type(16)));
typedef float v8f   __attribute__((ext_vector_type(8)));
typedef unsigned int u32x4 __attribute__((ext_vector_type(4)));
typedef int i32x4 __attribute__((ext_vector_type(4)));
typedef int i32x8 __attribute__((ext_vector_type(8)));

union FragAB { v16bf v; uint32_t w[8]; };

__device__ __forceinline__ uint16_t f2bf(float f) {
  uint32_t u = __float_as_uint(f);
  u += 0x7FFFu + ((u >> 16) & 1u);           // round-to-nearest-even
  return (uint16_t)(u >> 16);
}
__device__ __forceinline__ float bf2f(uint16_t h) {
  return __uint_as_float(((uint32_t)h) << 16);
}
__device__ __forceinline__ float siluf(float x) { return x / (1.f + __expf(-x)); }
__device__ __forceinline__ float ld_val(float v) { return v; }
__device__ __forceinline__ float ld_val(uint16_t v) { return bf2f(v); }

// ---------------------------------------------------------------------------
// Weight transpose + f32->bf16 convert (+K padding with zeros).
// W [K,N] f32 row-major  ->  Wt [N,Kpad] bf16 row-major.
// ---------------------------------------------------------------------------
__global__ void conv_wt_kernel(const float* __restrict__ W, uint16_t* __restrict__ Wt,
                               int K, int N, int Kpad) {
  int i = blockIdx.x * blockDim.x + threadIdx.x;
  if (i >= N * Kpad) return;
  int n = i / Kpad, k = i - n * Kpad;
  Wt[i] = (k < K) ? f2bf(W[(size_t)k * N + n]) : (uint16_t)0;
}

// Plain f32 -> bf16 convert.
__global__ void conv_bf_kernel(const float* __restrict__ X, uint16_t* __restrict__ Y, int n) {
  int i = blockIdx.x * blockDim.x + threadIdx.x;
  if (i < n) Y[i] = f2bf(X[i]);
}

// ---------------------------------------------------------------------------
// Query featurization: searchsorted + Fourier spatial enc + time/comp embed.
// Emits bf16 feature rows [N_Q, QFEAT] (cols 104..127 zero) and idx[n].
// ---------------------------------------------------------------------------
__global__ void qfeat_kernel(const float* __restrict__ xy, const float* __restrict__ tq,
                             const int* __restrict__ comp, const float* __restrict__ stime,
                             const float* __restrict__ tp_w, const float* __restrict__ tp_b,
                             const float* __restrict__ cemb,
                             uint16_t* __restrict__ F, int* __restrict__ idxArr) {
  int n = blockIdx.x * blockDim.x + threadIdx.x;
  if (n >= N_Q) return;
  float t = tq[n];
  int lo = 0, hi = TBINS;                      // upper_bound (side='right')
  while (lo < hi) { int mid = (lo + hi) >> 1; if (stime[mid] <= t) lo = mid + 1; else hi = mid; }
  int id = lo - 1; id = id < 0 ? 0 : (id > TBINS - 1 ? TBINS - 1 : id);
  idxArr[n] = id;
  float dt = fmaxf(t - stime[id], 0.f);

  uint16_t* f = F + (size_t)n * QFEAT;
  float x = xy[2 * n], y = xy[2 * n + 1];
  const float TWO_PI = 6.2831853071795864769f;
#pragma unroll
  for (int k = 0; k < 16; ++k) {
    float sx, cx, sy, cy;
    __sincosf(TWO_PI * x * (float)(k + 1), &sx, &cx);
    __sincosf(TWO_PI * y * (float)(k + 1), &sy, &cy);
    f[k] = f2bf(sx); f[16 + k] = f2bf(cx);
    f[32 + k] = f2bf(sy); f[48 + k] = f2bf(cy);
  }
#pragma unroll
  for (int j = 0; j < 32; ++j) f[64 + j] = f2bf(dt * tp_w[j] + tp_b[j]);
  int cc = comp[n];
#pragma unroll
  for (int j = 0; j < 8; ++j) f[96 + j] = f2bf(cemb[cc * 8 + j]);
#pragma unroll
  for (int j = 104; j < QFEAT; ++j) f[j] = 0;
}

// ---------------------------------------------------------------------------
// Tiled WMMA GEMM:  C[M,N] = act(A[M,Kd] @ Bt[N,Kd]^T + bias)
// Macro-tile 32x64, 8 waves, one 16x16 WMMA tile per wave.
// A macro-tile staged to LDS via TDM (tensor_load_to_lds).
// ---------------------------------------------------------------------------
template <int ACT_SILU, int OUT_BF16>
__global__ __launch_bounds__(256) void gemm_bf16_kernel(
    const uint16_t* __restrict__ A, const uint16_t* __restrict__ Bt,
    const float* __restrict__ bias, void* __restrict__ C,
    int M, int Ncol, int Kd) {
  __shared__ uint16_t lAs[32 * 384];                     // max Kd = 384 -> 24 KB
  const uint16_t* Atile = A + (size_t)blockIdx.x * 32 * Kd;

#if __has_builtin(__builtin_amdgcn_tensor_load_to_lds) && __has_builtin(__builtin_amdgcn_s_wait_tensorcnt)
  if (threadIdx.x < 32) {                                // one wave issues the DMA
    uint32_t lds_off = (uint32_t)(size_t)(&lAs[0]);
    uint64_t ga = (uint64_t)(size_t)Atile;
    uint32_t td0 = (uint32_t)Kd;                         // tensor dim0 (elems/row)
    uint32_t td1 = 32u;                                  // tensor dim1 (rows)
    u32x4 g0;
    g0[0] = 1u;                                          // count=1 (valid user D#)
    g0[1] = lds_off;                                     // lds_addr
    g0[2] = (uint32_t)ga;                                // global_addr[31:0]
    g0[3] = (uint32_t)((ga >> 32) & 0x1FFFFFFu) | (2u << 30);  // addr[56:32] | type=2
    i32x8 g1;
    g1[0] = (int)(1u << 16);                             // data_size = 2 bytes
    g1[1] = (int)((td0 & 0xFFFFu) << 16);                // tensor_dim0[15:0]
    g1[2] = (int)(((td0 >> 16) & 0xFFFFu) | ((td1 & 0xFFFFu) << 16));
    g1[3] = (int)(((td1 >> 16) & 0xFFFFu) | ((td0 & 0xFFFFu) << 16)); // tile_dim0 = Kd
    g1[4] = (int)32u;                                    // tile_dim1 = 32 rows
    g1[5] = (int)td0;                                    // tensor_dim0_stride = Kd
    g1[6] = 0; g1[7] = 0;
    i32x4 z4 = {0, 0, 0, 0};
    i32x8 z8 = {0, 0, 0, 0, 0, 0, 0, 0};
    __builtin_amdgcn_tensor_load_to_lds(g0, g1, z4, z4, z8, 0);
    __builtin_amdgcn_s_wait_tensorcnt(0);
  }
#else
  for (int i = threadIdx.x; i < 16 * Kd; i += 256)       // 32*Kd halves = 16*Kd dwords
    ((uint32_t*)lAs)[i] = ((const uint32_t*)Atile)[i];
#endif
  __syncthreads();

  const int wave = threadIdx.x >> 5, lane = threadIdx.x & 31;
  const int rt = wave & 1, ct = wave >> 1;               // 2 row-tiles x 4 col-tiles
  const int m0 = lane & 15, kh = lane >> 4;
  const int nCol = blockIdx.y * 64 + ct * 16 + m0;
  const uint16_t* brow = Bt + (size_t)nCol * Kd;
  const uint16_t* arow0 = &lAs[(rt * 16 + m0) * Kd];

  v8f acc = {};
  for (int kc = 0; kc < Kd; kc += 32) {
    __builtin_prefetch(brow + kc + 64, 0, 0);            // global_prefetch next chunk
    FragAB a, b;
    const uint16_t* bp = brow + kc + 8 * kh;
#pragma unroll
    for (int j = 0; j < 8; ++j) {
      // A fragment: 16-bit A layout, half-wave K-interleave {0..7|16..23}+8*kh
      int p0 = ((j < 4) ? (2 * j) : (16 + 2 * (j - 4))) + 8 * kh;
      a.w[j] = *(const uint32_t*)(arow0 + kc + p0);      // ds_load_b32
      // B fragment: lane = column nCol, same K-interleave {0..7|16..23}+8*kh as A
      b.w[j] = *(const uint32_t*)(bp + ((j < 4) ? (2 * j) : (16 + 2 * (j - 4))));           // global_load_b32
    }
    acc = __builtin_amdgcn_wmma_f32_16x16x32_bf16(false, a.v, false, b.v,
                                                  (short)0, acc, false, false);
  }

  const int rowBase = blockIdx.x * 32 + rt * 16;
  float bv = bias ? bias[nCol] : 0.f;
#pragma unroll
  for (int g = 0; g < 8; ++g) {                          // D layout: M=g+8*kh, N=lane&15
    int m = rowBase + g + 8 * kh;
    float xv = acc[g] + bv;
    if (ACT_SILU) xv = siluf(xv);
    size_t o = (size_t)m * Ncol + nCol;
    if (OUT_BF16) ((uint16_t*)C)[o] = f2bf(xv);
    else          ((float*)C)[o] = xv;
  }
}

// ---------------------------------------------------------------------------
// Row LayerNorm (width 256), one wave per row, bf16 output.
// ---------------------------------------------------------------------------
template <typename TIN>
__global__ __launch_bounds__(256) void ln256_kernel(const TIN* __restrict__ X,
                                                    const float* __restrict__ g,
                                                    const float* __restrict__ b,
                                                    uint16_t* __restrict__ Y, int rows) {
  int row = blockIdx.x * 8 + (threadIdx.x >> 5);
  int lane = threadIdx.x & 31;
  if (row >= rows) return;
  float x[8], s = 0.f, s2 = 0.f;
#pragma unroll
  for (int j = 0; j < 8; ++j) {
    float v = ld_val(X[(size_t)row * 256 + lane + 32 * j]);
    x[j] = v; s += v; s2 += v * v;
  }
  for (int o = 16; o; o >>= 1) { s += __shfl_xor(s, o, 32); s2 += __shfl_xor(s2, o, 32); }
  float mean = s * (1.f / 256.f);
  float var = s2 * (1.f / 256.f) - mean * mean;
  float inv = rsqrtf(var + 1e-5f);
#pragma unroll
  for (int j = 0; j < 8; ++j) {
    int col = lane + 32 * j;
    Y[(size_t)row * 256 + col] = f2bf((x[j] - mean) * inv * g[col] + b[col]);
  }
}

// ---------------------------------------------------------------------------
// Attention over K=48 sensors, one wave per query. rel-bias MLP is a constant
// (LN of a scalar) -> cancelled by softmax; only the locality decay survives.
// ---------------------------------------------------------------------------
__global__ __launch_bounds__(256) void attn_kernel(
    const float* __restrict__ qf, const float* __restrict__ kk, const float* __restrict__ vv,
    const int* __restrict__ idxArr, const float* __restrict__ xy,
    const float* __restrict__ sensor_pos, const float* __restrict__ log_decay,
    float* __restrict__ ctx) {
  int n = blockIdx.x * 8 + (threadIdx.x >> 5);
  int lane = threadIdx.x & 31;
  if (n >= N_Q) return;
  float qv[8];
#pragma unroll
  for (int j = 0; j < 8; ++j) qv[j] = qf[(size_t)n * 256 + lane + 32 * j];
  int t = idxArr[n];
  const float* kb = kk + (size_t)t * KSENS * 256;
  const float* vb = vv + (size_t)t * KSENS * 256;
  float decay = __expf(log_decay[0]);
  float qx = xy[2 * n], qy = xy[2 * n + 1];

  float sc[KSENS], mx = -1e30f;
  for (int k = 0; k < KSENS; ++k) {
    float d = 0.f;
#pragma unroll
    for (int j = 0; j < 8; ++j) d += qv[j] * kb[k * 256 + lane + 32 * j];
    for (int o = 16; o; o >>= 1) d += __shfl_xor(d, o, 32);
    float rx = qx - sensor_pos[2 * k]; rx -= rintf(rx);    // periodic wrap, L=1
    float ry = qy - sensor_pos[2 * k + 1]; ry -= rintf(ry);
    float r = sqrtf(rx * rx + ry * ry);
    float s = d * 0.0625f - decay * r;                     // 1/sqrt(256)
    sc[k] = s; mx = fmaxf(mx, s);
  }
  float sum = 0.f;
  for (int k = 0; k < KSENS; ++k) { sc[k] = __expf(sc[k] - mx); sum += sc[k]; }
  float inv = 1.f / sum;
#pragma unroll
  for (int j = 0; j < 8; ++j) {
    float c = 0.f;
    for (int k = 0; k < KSENS; ++k) c += sc[k] * vb[k * 256 + lane + 32 * j];
    ctx[(size_t)n * 256 + lane + 32 * j] = c * inv;
  }
}

// Residual add, bf16 output (ctx2 = ctx + delta).
__global__ void add_bf_kernel(const float* __restrict__ a, const float* __restrict__ d,
                              uint16_t* __restrict__ y, int n) {
  int i = blockIdx.x * blockDim.x + threadIdx.x;
  if (i < n) y[i] = f2bf(a[i] + d[i]);
}

// ---------------------------------------------------------------------------
// Final fusion: rank-128 dot of component-selected trunk/branch bases.
// ---------------------------------------------------------------------------
__global__ __launch_bounds__(256) void final_kernel(
    const float* __restrict__ tb, const float* __restrict__ bb, const int* __restrict__ comp,
    const float* __restrict__ log_temp, const float* __restrict__ cscale,
    const float* __restrict__ cbias, float* __restrict__ out) {
  int n = blockIdx.x * 8 + (threadIdx.x >> 5);
  int lane = threadIdx.x & 31;
  if (n >= N_Q) return;
  int cc = comp[n];
  const float* tr = tb + (size_t)n * 384 + cc * RANKD;
  const float* br = bb + (size_t)n * 384 + cc * RANKD;
  float d = 0.f;
#pragma unroll
  for (int j = 0; j < 4; ++j) d += tr[lane + 32 * j] * br[lane + 32 * j];
  for (int o = 16; o; o >>= 1) d += __shfl_xor(d, o, 32);
  if (lane == 0) out[n] = d * __expf(log_temp[0]) * cscale[cc] + cbias[cc];
}

// ---------------------------------------------------------------------------
extern "C" void kernel_launch(void* const* d_in, const int* in_sizes, int n_in,
                              void* d_out, int out_size, void* d_ws, size_t ws_size,
                              hipStream_t stream) {
  (void)in_sizes; (void)n_in; (void)out_size; (void)ws_size;
  const float* xy          = (const float*)d_in[0];
  const float* t_q         = (const float*)d_in[1];
  const int*   comp        = (const int*)d_in[2];
  const float* h_states    = (const float*)d_in[3];
  const float* sensor_time = (const float*)d_in[4];
  const float* sensor_pos  = (const float*)d_in[5];
  const float* p_time_w = (const float*)d_in[6];
  const float* p_time_b = (const float*)d_in[7];
  const float* p_cemb   = (const float*)d_in[8];
  const float* p_tin_w  = (const float*)d_in[9];
  const float* p_tin_b  = (const float*)d_in[10];
  const float* p_bn_g   = (const float*)d_in[11];
  const float* p_bn_b   = (const float*)d_in[12];
  const float* p_tok_w  = (const float*)d_in[13];
  const float* p_tok_b  = (const float*)d_in[14];
  const float* p_q_w    = (const float*)d_in[15];
  const float* p_q_b    = (const float*)d_in[16];
  const float* p_k_w    = (const float*)d_in[17];
  const float* p_k_b    = (const float*)d_in[18];
  const float* p_v_w    = (const float*)d_in[19];
  const float* p_v_b    = (const float*)d_in[20];
  // d_in[21..26]: rp_* params — analytically a no-op (see header comment).
  const float* p_bc_g   = (const float*)d_in[27];
  const float* p_bc_b   = (const float*)d_in[28];
  const float* p_bc1_w  = (const float*)d_in[29];
  const float* p_bc1_b  = (const float*)d_in[30];
  const float* p_bc2_w  = (const float*)d_in[31];
  const float* p_bc2_b  = (const float*)d_in[32];
  const float* p_tout_w = (const float*)d_in[33];
  const float* p_tout_b = (const float*)d_in[34];
  const float* p_bp_w   = (const float*)d_in[35];
  const float* p_bp_b   = (const float*)d_in[36];
  const float* p_ltemp  = (const float*)d_in[37];
  const float* p_cscale = (const float*)d_in[38];
  const float* p_cbias  = (const float*)d_in[39];
  const float* p_ldecay = (const float*)d_in[40];

  // ---- workspace carve ----
  size_t off = 0;
  auto carve = [&](size_t bytes) -> void* {
    void* p = (char*)d_ws + off;
    off += (bytes + 255) & ~(size_t)255;
    return p;
  };
  const int MTK = TBINS * KSENS;  // 9600
  uint16_t* w_tok_t  = (uint16_t*)carve((size_t)HDIM * DMODEL * 2);   // [256,384]
  uint16_t* w_k_t    = (uint16_t*)carve((size_t)HDIM * HDIM * 2);
  uint16_t* w_v_t    = (uint16_t*)carve((size_t)HDIM * HDIM * 2);
  uint16_t* w_q_t    = (uint16_t*)carve((size_t)HDIM * HDIM * 2);
  uint16_t* w_bc1_t  = (uint16_t*)carve((size_t)HDIM * HDIM * 2);
  uint16_t* w_bc2_t  = (uint16_t*)carve((size_t)HDIM * HDIM * 2);
  uint16_t* w_tin_t  = (uint16_t*)carve((size_t)HDIM * QFEAT * 2);    // [256,128]
  uint16_t* w_tout_t = (uint16_t*)carve((size_t)384 * HDIM * 2);      // [384,256]
  uint16_t* w_bp_t   = (uint16_t*)carve((size_t)384 * HDIM * 2);
  uint16_t* h_bf     = (uint16_t*)carve((size_t)MTK * DMODEL * 2);
  uint16_t* Fq       = (uint16_t*)carve((size_t)N_Q * QFEAT * 2);
  uint16_t* tok_bf   = (uint16_t*)carve((size_t)MTK * HDIM * 2);
  uint16_t* tf_bf    = (uint16_t*)carve((size_t)N_Q * HDIM * 2);
  uint16_t* bq_bf    = (uint16_t*)carve((size_t)N_Q * HDIM * 2);
  uint16_t* cn_bf    = (uint16_t*)carve((size_t)N_Q * HDIM * 2);
  uint16_t* h2_bf    = (uint16_t*)carve((size_t)N_Q * HDIM * 2);
  uint16_t* ctx2_bf  = (uint16_t*)carve((size_t)N_Q * HDIM * 2);
  float*    kkf      = (float*)carve((size_t)MTK * HDIM * 4);
  float*    vvf      = (float*)carve((size_t)MTK * HDIM * 4);
  float*    qff      = (float*)carve((size_t)N_Q * HDIM * 4);
  float*    tbasis   = (float*)carve((size_t)N_Q * 384 * 4);
  float*    ctxf     = (float*)carve((size_t)N_Q * HDIM * 4);
  float*    deltaf   = (float*)carve((size_t)N_Q * HDIM * 4);
  float*    bbasis   = (float*)carve((size_t)N_Q * 384 * 4);
  int*      idxArr   = (int*)carve((size_t)N_Q * 4);

  auto blks = [](int total) { return (total + 255) / 256; };

  // ---- weight prep (bf16, transposed, K padded) ----
  conv_wt_kernel<<<blks(HDIM * DMODEL), 256, 0, stream>>>(p_tok_w, w_tok_t, DMODEL, HDIM, DMODEL);
  conv_wt_kernel<<<blks(HDIM * HDIM), 256, 0, stream>>>(p_k_w,  w_k_t,  HDIM, HDIM, HDIM);
  conv_wt_kernel<<<blks(HDIM * HDIM), 256, 0, stream>>>(p_v_w,  w_v_t,  HDIM, HDIM, HDIM);
  conv_wt_kernel<<<blks(HDIM * HDIM), 256, 0, stream>>>(p_q_w,  w_q_t,  HDIM, HDIM, HDIM);
  conv_wt_kernel<<<blks(HDIM * HDIM), 256, 0, stream>>>(p_bc1_w, w_bc1_t, HDIM, HDIM, HDIM);
  conv_wt_kernel<<<blks(HDIM * HDIM), 256, 0, stream>>>(p_bc2_w, w_bc2_t, HDIM, HDIM, HDIM);
  conv_wt_kernel<<<blks(HDIM * QFEAT), 256, 0, stream>>>(p_tin_w, w_tin_t, 104, HDIM, QFEAT);
  conv_wt_kernel<<<blks(384 * HDIM), 256, 0, stream>>>(p_tout_w, w_tout_t, HDIM, 384, HDIM);
  conv_wt_kernel<<<blks(384 * HDIM), 256, 0, stream>>>(p_bp_w,   w_bp_t,   HDIM, 384, HDIM);
  conv_bf_kernel<<<blks(MTK * DMODEL), 256, 0, stream>>>(h_states, h_bf, MTK * DMODEL);

  // ---- query featurization ----
  qfeat_kernel<<<blks(N_Q), 256, 0, stream>>>(xy, t_q, comp, sensor_time,
                                              p_time_w, p_time_b, p_cemb, Fq, idxArr);

  // ---- per-time-bin token / K / V projections (WMMA) ----
  gemm_bf16_kernel<0, 1><<<dim3(MTK / 32, HDIM / 64), 256, 0, stream>>>(
      h_bf, w_tok_t, p_tok_b, tok_bf, MTK, HDIM, DMODEL);
  gemm_bf16_kernel<0, 0><<<dim3(MTK / 32, HDIM / 64), 256, 0, stream>>>(
      tok_bf, w_k_t, p_k_b, kkf, MTK, HDIM, HDIM);
  gemm_bf16_kernel<0, 0><<<dim3(MTK / 32, HDIM / 64), 256, 0, stream>>>(
      tok_bf, w_v_t, p_v_b, vvf, MTK, HDIM, HDIM);

  // ---- trunk path (WMMA) ----
  gemm_bf16_kernel<1, 1><<<dim3(N_Q / 32, HDIM / 64), 256, 0, stream>>>(
      Fq, w_tin_t, p_tin_b, tf_bf, N_Q, HDIM, QFEAT);                  // trunk_feat (SiLU)
  ln256_kernel<uint16_t><<<N_Q / 8, 256, 0, stream>>>(tf_bf, p_bn_g, p_bn_b, bq_bf, N_Q);
  gemm_bf16_kernel<0, 0><<<dim3(N_Q / 32, HDIM / 64), 256, 0, stream>>>(
      bq_bf, w_q_t, p_q_b, qff, N_Q, HDIM, HDIM);                      // q
  gemm_bf16_kernel<0, 0><<<dim3(N_Q / 32, 384 / 64), 256, 0, stream>>>(
      tf_bf, w_tout_t, p_tout_b, tbasis, N_Q, 384, HDIM);              // trunk_basis

  // ---- attention ----
  attn_kernel<<<N_Q / 8, 256, 0, stream>>>(qff, kkf, vvf, idxArr, xy,
                                           sensor_pos, p_ldecay, ctxf);

  // ---- branch context MLP + projection (WMMA) ----
  ln256_kernel<float><<<N_Q / 8, 256, 0, stream>>>(ctxf, p_bc_g, p_bc_b, cn_bf, N_Q);
  gemm_bf16_kernel<1, 1><<<dim3(N_Q / 32, HDIM / 64), 256, 0, stream>>>(
      cn_bf, w_bc1_t, p_bc1_b, h2_bf, N_Q, HDIM, HDIM);                // h2 (SiLU)
  gemm_bf16_kernel<0, 0><<<dim3(N_Q / 32, HDIM / 64), 256, 0, stream>>>(
      h2_bf, w_bc2_t, p_bc2_b, deltaf, N_Q, HDIM, HDIM);               // residual delta
  add_bf_kernel<<<blks(N_Q * HDIM), 256, 0, stream>>>(ctxf, deltaf, ctx2_bf, N_Q * HDIM);
  gemm_bf16_kernel<0, 0><<<dim3(N_Q / 32, 384 / 64), 256, 0, stream>>>(
      ctx2_bf, w_bp_t, p_bp_b, bbasis, N_Q, 384, HDIM);                // branch_basis

  // ---- final fusion ----
  final_kernel<<<N_Q / 8, 256, 0, stream>>>(tbasis, bbasis, comp,
                                            p_ltemp, p_cscale, p_cbias, (float*)d_out);
}
